// QuantLlamaAttention_21878563406209
// MI455X (gfx1250) — matrix operand order repaired and checked by reading, not compiled, not run on hardware
//
#include <hip/hip_runtime.h>

#define B_  2
#define S_  2048
#define H_  4096
#define NH_ 32
#define HD_ 128

typedef __bf16 bf16;
typedef __attribute__((ext_vector_type(16))) __bf16 v16bf;
typedef __attribute__((ext_vector_type(8)))  __bf16 v8bf;
typedef __attribute__((ext_vector_type(8)))  float  v8f;
typedef __attribute__((ext_vector_type(4)))  int    v4i;

// gfx1250 async global->LDS path (ASYNCcnt). Fallback: load + ds_store.
#if defined(__AMDGCN__) &&                                                   \
    __has_builtin(__builtin_amdgcn_global_load_async_to_lds_b128) &&         \
    __has_builtin(__builtin_amdgcn_s_wait_asynccnt)
#define USE_ASYNC_LDS 1
#else
#define USE_ASYNC_LDS 0
#endif

// ---------------------------------------------------------------------------
// WMMA: D = A(16x32 bf16) * B(32x16 bf16) + C(16x16 f32)
// ---------------------------------------------------------------------------
__device__ __forceinline__ v8f wmma_bf(v16bf a, v16bf b, v8f c) {
  return __builtin_amdgcn_wmma_f32_16x16x32_bf16(
      /*neg_a=*/false, a, /*neg_b=*/false, b,
      /*c_mod=*/(short)0, c, /*reuse_a=*/false, /*reuse_b=*/false);
}

// A-fragment (16x32, 16-bit): lane r = M row, hi = lane>>4.
// elements j -> K = hi*16 + j
__device__ __forceinline__ v16bf ldA(const bf16* base, int ld, int r, int hi) {
  const bf16* p = base + (size_t)r * ld + hi * 16;
  union { v16bf v; v8bf h[2]; } u;
  u.h[0] = *(const v8bf*)(p);
  u.h[1] = *(const v8bf*)(p + 8);
  return u.v;
}

// B-fragment (32x16, 16-bit): lane c = N col, elements j -> K = hi*16 + j.
// base must be [N][K] row-major (i.e. B transposed).
__device__ __forceinline__ v16bf ldB(const bf16* base, int ld, int c, int hi) {
  const bf16* p = base + (size_t)c * ld + hi * 16;
  union { v16bf v; v8bf h[2]; } u;
  u.h[0] = *(const v8bf*)(p);
  u.h[1] = *(const v8bf*)(p + 8);
  return u.v;
}

// Stage 16 bf16 (32B) from global to LDS.
__device__ __forceinline__ void stage16(const bf16* g, bf16* l) {
#if USE_ASYNC_LDS
  __builtin_amdgcn_global_load_async_to_lds_b128((v4i*)g, (v4i*)l, 0, 0);
  __builtin_amdgcn_global_load_async_to_lds_b128((v4i*)(g + 8), (v4i*)(l + 8),
                                                 0, 0);
#else
  *(v8bf*)(l)     = *(const v8bf*)(g);
  *(v8bf*)(l + 8) = *(const v8bf*)(g + 8);
#endif
}

__device__ __forceinline__ void wait_stage() {
#if USE_ASYNC_LDS
  __builtin_amdgcn_s_wait_asynccnt(0);
#endif
}

// ---------------------------------------------------------------------------
// fp32 -> bf16 elementwise
// ---------------------------------------------------------------------------
__global__ __launch_bounds__(256) void cvt_bf16(const float* __restrict__ x,
                                                bf16* __restrict__ y, int n) {
  int i = blockIdx.x * 256 + threadIdx.x;
  if (i < n) y[i] = (bf16)x[i];
}

// W[R][C] f32  ->  Wt[C][R] bf16 (LDS-tiled transpose)
__global__ __launch_bounds__(256) void transpose_cvt(const float* __restrict__ W,
                                                     bf16* __restrict__ Wt,
                                                     int R, int C) {
  __shared__ bf16 tile[32][33];
  const int cb = blockIdx.x * 32, rb = blockIdx.y * 32;
  const int tx = threadIdx.x & 31, ty = threadIdx.x >> 5;
#pragma unroll
  for (int i = 0; i < 32; i += 8)
    tile[ty + i][tx] = (bf16)W[(size_t)(rb + ty + i) * C + cb + tx];
  __syncthreads();
#pragma unroll
  for (int i = 0; i < 32; i += 8)
    Wt[(size_t)(cb + ty + i) * R + rb + tx] = tile[tx][ty + i];
}

// ---------------------------------------------------------------------------
// GEMM: C[M][N] f32 = A[M][K] bf16 * Bt[N][K]^T bf16
// block tile 128x128, 8 waves (4 M x 2 N), wave tile 32x64, K-step 32,
// double-buffered LDS (one barrier per K-step, staging overlaps compute).
// ---------------------------------------------------------------------------
__global__ __launch_bounds__(256)
void gemm_bf16f32(const bf16* __restrict__ A, const bf16* __restrict__ Bt,
                  float* __restrict__ C, int M, int N, int K) {
  constexpr int TM = 128, TN = 128, TK = 32, LDSS = 40;  // 40 = TK + 8 pad
  __shared__ bf16 sA[2][TM * LDSS];
  __shared__ bf16 sB[2][TN * LDSS];
  (void)M; (void)TN;

  const int m0 = blockIdx.y * TM;
  const int n0 = blockIdx.x * TN;
  const int tid = threadIdx.x;
  const int wave = tid >> 5;
  const int wm = wave >> 1, wn = wave & 1;
  const int lane = tid & 31;
  const int r = lane & 15, hi = lane >> 4;

  // staging map: each thread owns 16 bf16 of one row, for both tiles
  const int row = tid >> 1, col = (tid & 1) * 16;
  const bf16* gA = A + (size_t)(m0 + row) * K + col;
  const bf16* gB = Bt + (size_t)(n0 + row) * K + col;
  const int loff = row * LDSS + col;

  v8f acc[2][4] = {};

  stage16(gA, &sA[0][loff]);
  stage16(gB, &sB[0][loff]);

  int p = 0;
  for (int k0 = 0; k0 < K; k0 += TK) {
    wait_stage();
    __syncthreads();
    if (k0 + TK < K) {
      stage16(gA + k0 + TK, &sA[p ^ 1][loff]);
      stage16(gB + k0 + TK, &sB[p ^ 1][loff]);
    }
    v16bf a0 = ldA(&sA[p][(wm * 32 +  0) * LDSS], LDSS, r, hi);
    v16bf a1 = ldA(&sA[p][(wm * 32 + 16) * LDSS], LDSS, r, hi);
#pragma unroll
    for (int j = 0; j < 4; ++j) {
      v16bf bj = ldB(&sB[p][(wn * 64 + j * 16) * LDSS], LDSS, r, hi);
      acc[0][j] = wmma_bf(a0, bj, acc[0][j]);
      acc[1][j] = wmma_bf(a1, bj, acc[1][j]);
    }
    p ^= 1;
  }

#pragma unroll
  for (int i = 0; i < 2; ++i)
#pragma unroll
    for (int j = 0; j < 4; ++j)
#pragma unroll
      for (int v = 0; v < 8; ++v) {
        int rw = m0 + wm * 32 + i * 16 + hi * 8 + v;
        int cl = n0 + wn * 64 + j * 16 + r;
        C[(size_t)rw * N + cl] = acc[i][j][v];
      }
}

// ---------------------------------------------------------------------------
// RoPE + repack: qkv f32 [B*S][3H] -> Qh/Kh bf16 [B*NH][S][HD] (roped),
// Vt bf16 [B*NH][HD][S] (transposed for the O^T = V^T P^T GEMM).
// ---------------------------------------------------------------------------
__global__ __launch_bounds__(256)
void rope_repack(const float* __restrict__ qkv, const int* __restrict__ pids,
                 bf16* __restrict__ Qh, bf16* __restrict__ Kh,
                 bf16* __restrict__ Vt) {
  const size_t idx = (size_t)blockIdx.x * 256 + threadIdx.x;
  const int i = (int)(idx & 63);          // 0..63 = half of HD
  const size_t t = idx >> 6;              // (b*S + s)*NH + h
  const int h = (int)(t % NH_);
  const size_t bs = t / NH_;
  const int s = (int)(bs % S_);
  const int b = (int)(bs / S_);

  const float* base = qkv + bs * (size_t)(3 * H_);
  const int col = h * HD_;
  float q1 = base[col + i],           q2 = base[col + 64 + i];
  float k1 = base[H_ + col + i],      k2 = base[H_ + col + 64 + i];
  float v1 = base[2 * H_ + col + i],  v2 = base[2 * H_ + col + 64 + i];

  float pos = (float)pids[bs];
  float freq = pos * __expf(((float)(2 * i) / (float)HD_) * -9.210340371976184f);
  float sn, cs;
  __sincosf(freq, &sn, &cs);

  const size_t head = (size_t)(b * NH_ + h);
  bf16* Qp = Qh + (head * S_ + (size_t)s) * HD_;
  bf16* Kp = Kh + (head * S_ + (size_t)s) * HD_;
  Qp[i]      = (bf16)(q1 * cs - q2 * sn);
  Qp[64 + i] = (bf16)(q2 * cs + q1 * sn);
  Kp[i]      = (bf16)(k1 * cs - k2 * sn);
  Kp[64 + i] = (bf16)(k2 * cs + k1 * sn);
  bf16* Vp = Vt + head * (size_t)(HD_ * S_);
  Vp[(size_t)i * S_ + s]        = (bf16)v1;
  Vp[(size_t)(64 + i) * S_ + s] = (bf16)v2;
}

// ---------------------------------------------------------------------------
// Causal flash attention, one wave = 16 query rows.
// Computes S^T = K Q^T (softmax stats per-lane + one shfl_xor(16)),
// then O^T = V^T P^T. Writes attn bf16 [B][S][H] (h*HD+d packing).
// ---------------------------------------------------------------------------
__global__ __launch_bounds__(256)
void flash_attn(const bf16* __restrict__ Qh, const bf16* __restrict__ Kh,
                const bf16* __restrict__ Vt, bf16* __restrict__ attn) {
  const int bh = blockIdx.y;                 // b*NH + h
  const int b = bh / NH_, h = bh % NH_;
  const int wave = threadIdx.x >> 5;
  const int lane = threadIdx.x & 31;
  const int r = lane & 15, hi = lane >> 4;
  const int q0 = blockIdx.x * 128 + wave * 16;
  const float scale = 0.08838834764831845f;  // 1/sqrt(128)

  const bf16* Q = Qh + (size_t)bh * S_ * HD_;
  const bf16* K = Kh + (size_t)bh * S_ * HD_;
  const bf16* V = Vt + (size_t)bh * HD_ * S_;   // [HD][S]

  // Q loaded B-style: gives Q^T as the WMMA B operand. lane r = query col.
  v16bf qf[4];
#pragma unroll
  for (int f = 0; f < 4; ++f)
    qf[f] = ldB(Q + (size_t)q0 * HD_ + f * 32, HD_, r, hi);

  v8f o[8] = {};                 // O^T frags: frag f = d rows 16f..16f+15
  float m = -1e30f, l = 0.f;     // per-query stats (lane's query = q0+r)
  const int qcol = q0 + r;

  for (int kv0 = 0; kv0 < q0 + 16; kv0 += 32) {
    v8f st0 = {}, st1 = {};
#pragma unroll
    for (int f = 0; f < 4; ++f) {
      v16bf kf0 = ldA(K + (size_t)kv0 * HD_ + f * 32, HD_, r, hi);
      v16bf kf1 = ldA(K + (size_t)(kv0 + 16) * HD_ + f * 32, HD_, r, hi);
      st0 = wmma_bf(kf0, qf[f], st0);
      st1 = wmma_bf(kf1, qf[f], st1);
    }
#pragma unroll
    for (int i = 0; i < 8; ++i) {
      int kvA = kv0 + hi * 8 + i;
      int kvB = kv0 + 16 + hi * 8 + i;
      st0[i] = (kvA <= qcol) ? st0[i] * scale : -1e30f;
      st1[i] = (kvB <= qcol) ? st1[i] * scale : -1e30f;
    }
    float tmax = -1e30f;
#pragma unroll
    for (int i = 0; i < 8; ++i) tmax = fmaxf(tmax, fmaxf(st0[i], st1[i]));
    tmax = fmaxf(tmax, __shfl_xor(tmax, 16, 32));
    float mnew = fmaxf(m, tmax);
    float corr = __expf(m - mnew);
    float psum = 0.f;
#pragma unroll
    for (int i = 0; i < 8; ++i) {
      st0[i] = __expf(st0[i] - mnew);
      st1[i] = __expf(st1[i] - mnew);
      psum += st0[i] + st1[i];
    }
    psum += __shfl_xor(psum, 16, 32);
    l = l * corr + psum;
    m = mnew;
#pragma unroll
    for (int f = 0; f < 8; ++f) o[f] *= corr;

    // P^T -> B-fragment: lane hi=0 wants kv 0..15, hi=1 wants kv 16..31;
    // exchange the missing 8 kv values with the opposite half-wave.
    union { v16bf v; bf16 e[16]; } pb;
#pragma unroll
    for (int i = 0; i < 8; ++i) {
      float e0 = st0[i], e1 = st1[i];
      float e0x = __shfl_xor(e0, 16, 32);
      float e1x = __shfl_xor(e1, 16, 32);
      pb.e[i]     = (bf16)(hi ? e1x : e0);
      pb.e[8 + i] = (bf16)(hi ? e1 : e0x);
    }
#pragma unroll
    for (int f = 0; f < 8; ++f) {
      v16bf vf = ldA(V + (size_t)(f * 16) * S_ + kv0, S_, r, hi);
      o[f] = wmma_bf(vf, pb.v, o[f]);
    }
  }

  const float linv = 1.f / l;
  bf16* outp = attn + ((size_t)b * S_ + qcol) * H_ + h * HD_;
#pragma unroll
  for (int f = 0; f < 8; ++f)
#pragma unroll
    for (int i = 0; i < 8; ++i)
      outp[f * 16 + hi * 8 + i] = (bf16)(o[f][i] * linv);
}

// ---------------------------------------------------------------------------
extern "C" void kernel_launch(void* const* d_in, const int* in_sizes, int n_in,
                              void* d_out, int out_size, void* d_ws,
                              size_t ws_size, hipStream_t stream) {
  (void)in_sizes; (void)n_in; (void)out_size; (void)ws_size;
  const float* hidden = (const float*)d_in[0];
  const float* w_qkv  = (const float*)d_in[1];
  const float* w_o    = (const float*)d_in[2];
  const int*   pids   = (const int*)d_in[3];
  float* out = (float*)d_out;

  char* w = (char*)d_ws;
  auto carve = [&](size_t bytes) {
    char* p = w;
    w += (bytes + 255) & ~(size_t)255;
    return (void*)p;
  };
  bf16*  hid_bf = (bf16*)carve((size_t)B_ * S_ * H_ * 2);
  bf16*  wqkvT  = (bf16*)carve((size_t)H_ * 3 * H_ * 2);
  bf16*  woT    = (bf16*)carve((size_t)H_ * H_ * 2);
  float* qkvf   = (float*)carve((size_t)B_ * S_ * 3 * H_ * 4);
  bf16*  Qhp    = (bf16*)carve((size_t)B_ * S_ * H_ * 2);
  bf16*  Khp    = (bf16*)carve((size_t)B_ * S_ * H_ * 2);
  bf16*  Vtp    = (bf16*)carve((size_t)B_ * S_ * H_ * 2);
  bf16*  attnb  = (bf16*)carve((size_t)B_ * S_ * H_ * 2);

  const int nh = B_ * S_ * H_;
  cvt_bf16<<<(nh + 255) / 256, 256, 0, stream>>>(hidden, hid_bf, nh);
  transpose_cvt<<<dim3(3 * H_ / 32, H_ / 32), 256, 0, stream>>>(w_qkv, wqkvT, H_, 3 * H_);
  transpose_cvt<<<dim3(H_ / 32, H_ / 32), 256, 0, stream>>>(w_o, woT, H_, H_);
  gemm_bf16f32<<<dim3(3 * H_ / 128, B_ * S_ / 128), 256, 0, stream>>>(
      hid_bf, wqkvT, qkvf, B_ * S_, 3 * H_, H_);
  rope_repack<<<(B_ * S_ * NH_ * 64) / 256, 256, 0, stream>>>(qkvf, pids, Qhp, Khp, Vtp);
  flash_attn<<<dim3(S_ / 128, B_ * NH_), 256, 0, stream>>>(Qhp, Khp, Vtp, attnb);
  gemm_bf16f32<<<dim3(H_ / 128, B_ * S_ / 128), 256, 0, stream>>>(
      attnb, woT, out, B_ * S_, H_, H_);
}
